// ProtoNCELoss_80650895884988
// MI455X (gfx1250) — matrix operand order repaired and checked by reading, not compiled, not run on hardware
//
#include <hip/hip_runtime.h>

// ---------------------------------------------------------------------------
// ProtoNCE loss, MI455X (gfx1250), wave32 + WMMA f16 path.
//   features (B=262144, D=256) f32, prototypes (K=64, D=256) f32, labels (B,) int
//   loss = mean_b -log((exp(sim[b,l_b]) + eps)/(sum_k exp(sim[b,k]) + eps) + eps)
//   sim[b,k] = (f_b . p_hat_k) / (||f_b|| * tau)   -> f16 WMMA GEMM, f32 accum
// Memory-bound floor: 268 MB features @ 23.3 TB/s ~= 11.5 us; matrix pipe needed.
// ---------------------------------------------------------------------------

typedef _Float16 v16h __attribute__((ext_vector_type(16)));
typedef _Float16 v8h  __attribute__((ext_vector_type(8)));
typedef _Float16 v4h  __attribute__((ext_vector_type(4)));
typedef float    v8f  __attribute__((ext_vector_type(8)));
typedef float    v4f  __attribute__((ext_vector_type(4)));

#define B_ROWS        262144
#define DDIM          256
#define KPROTO        64
#define LDA           264      // padded halves per A row (avoids LDS bank conflicts)
#define LDB           264      // padded halves per B row
#define WAVES         8
#define ROWS_PER_BLOCK 128     // 8 waves x 16 rows
#define NBLOCKS       (B_ROWS / ROWS_PER_BLOCK)   // 2048
#define TAU_INV       10.0f
#define EPS_F         1e-8f
#define LOG2E_F       1.44269504088896f
#define LN2_F         0.69314718055995f

// ---------------- kernel 1: normalize prototypes to f16 --------------------
__global__ void __launch_bounds__(256) proto_norm_kernel(
    const float* __restrict__ proto, _Float16* __restrict__ pnorm)
{
    const int k = blockIdx.x;       // 64 blocks
    const int d = threadIdx.x;      // 256 threads = D
    float v  = proto[k * DDIM + d];
    float ss = v * v;
#pragma unroll
    for (int m = 1; m < 32; m <<= 1) ss += __shfl_xor(ss, m, 32);
    __shared__ float red[8];
    const int wv = threadIdx.x >> 5, ln = threadIdx.x & 31;
    if (ln == 0) red[wv] = ss;
    __syncthreads();
    float tot = 0.f;
#pragma unroll
    for (int w = 0; w < 8; ++w) tot += red[w];
    const float rn = 1.0f / fmaxf(sqrtf(tot), 1e-12f);
    pnorm[k * DDIM + d] = (_Float16)(v * rn);
}

// ---------------- kernel 2: fused GEMM + softmax-NLL partials ---------------
__global__ void __launch_bounds__(256) protonce_main_kernel(
    const float* __restrict__ feats,
    const _Float16* __restrict__ pnorm,
    const int* __restrict__ labels,
    float* __restrict__ partials)
{
    __shared__ __align__(16) _Float16 shB[KPROTO * LDB];        // 33792 B
    __shared__ __align__(16) _Float16 shA[WAVES * 16 * LDA];    // 67584 B
    __shared__ float shNorm[WAVES * 16];
    __shared__ float shWave[WAVES];

    const int tid  = threadIdx.x;
    const int wave = tid >> 5;
    const int lane = tid & 31;

    // ---- stage normalized prototypes (f16, padded stride) into LDS ----
    {
        const uint4* src = (const uint4*)pnorm;   // 8 halves / uint4
#pragma unroll
        for (int it = 0; it < 8; ++it) {
            const int idx = it * 256 + tid;       // 0..2047 chunks
            const int h   = idx << 3;
            const int row = h >> 8;               // /256
            const int col = h & 255;
            *(uint4*)&shB[row * LDB + col] = src[idx];
        }
    }

    // ---- stage 16 feature rows per wave as f16; fold row sumsq in fp32 ----
    {
        const int rt   = lane >> 1;               // row in wave tile 0..15
        const int half = lane & 1;                // which 128-col half
        const int rowg = blockIdx.x * ROWS_PER_BLOCK + wave * 16 + rt;
        const float* fsrc = feats + (size_t)rowg * DDIM + half * 128;
        _Float16* adst = &shA[(wave * 16 + rt) * LDA + half * 128];
        float ss = 0.f;
#pragma unroll
        for (int i = 0; i < 32; ++i) {
            v4f v = *(const v4f*)(fsrc + i * 4);
            ss += v[0]*v[0] + v[1]*v[1] + v[2]*v[2] + v[3]*v[3];
            v4h hv = { (_Float16)v[0], (_Float16)v[1],
                       (_Float16)v[2], (_Float16)v[3] };
            *(v4h*)(adst + i * 4) = hv;
        }
        ss += __shfl_xor(ss, 1, 32);              // combine the two half-rows
        const float rn = 1.0f / fmaxf(sqrtf(ss), 1e-12f);
        if (half == 0) shNorm[wave * 16 + rt] = rn;
    }
    __syncthreads();

    // ---- WMMA: C(16x64) = A(16x256,f16) x B(256x64,f16), 32 wmma ops/wave ----
    v8f acc[4];
#pragma unroll
    for (int nt = 0; nt < 4; ++nt)
#pragma unroll
        for (int j = 0; j < 8; ++j) acc[nt][j] = 0.f;

    // A frag: lane<16 row=lane holds K {kb+0..7, kb+16..23}; lane>=16 the other split
    const _Float16* abase =
        &shA[(wave * 16 + (lane & 15)) * LDA + ((lane >> 4) << 3)];
    // B frag: lane<16 N=lane holds K {kb+0..7, kb+16..23}; lane>=16 the other split
    const _Float16* bbase =
        &shB[(lane & 15) * LDB + ((lane >> 4) << 3)];

#pragma unroll
    for (int kk = 0; kk < 8; ++kk) {
        const int kb = kk * 32;
        v8h a0 = *(const v8h*)(abase + kb);
        v8h a1 = *(const v8h*)(abase + kb + 16);
        v16h a = __builtin_shufflevector(a0, a1,
                 0,1,2,3,4,5,6,7,8,9,10,11,12,13,14,15);
#pragma unroll
        for (int nt = 0; nt < 4; ++nt) {
            const _Float16* bp = bbase + nt * 16 * LDB + kb;
            v8h b0 = *(const v8h*)(bp);
            v8h b1 = *(const v8h*)(bp + 16);
            v16h b = __builtin_shufflevector(b0, b1,
                     0,1,2,3,4,5,6,7,8,9,10,11,12,13,14,15);
            acc[nt] = __builtin_amdgcn_wmma_f32_16x16x32_f16(
                false, a, false, b, (short)0, acc[nt], false, false);
        }
    }

    // ---- epilogue: exp, denominator + label-numerator, per-row loss ----
    // C layout: VGPR j -> row = j + 8*(lane>=16); N = (lane&15) within n-tile
    const int rsel = (lane >> 4) << 3;
    float myrn[8];
    int   lbl[8];
#pragma unroll
    for (int j = 0; j < 8; ++j) {
        const int r = rsel + j;
        myrn[j] = shNorm[wave * 16 + r];
        lbl[j]  = labels[blockIdx.x * ROWS_PER_BLOCK + wave * 16 + r];
    }
    float den[8], num[8];
#pragma unroll
    for (int j = 0; j < 8; ++j) { den[j] = 0.f; num[j] = 0.f; }
#pragma unroll
    for (int nt = 0; nt < 4; ++nt) {
        const int proto = nt * 16 + (lane & 15);
#pragma unroll
        for (int j = 0; j < 8; ++j) {
            const float e = __builtin_amdgcn_exp2f(
                acc[nt][j] * myrn[j] * (TAU_INV * LOG2E_F));
            den[j] += e;
            num[j] += (proto == lbl[j]) ? e : 0.f;
        }
    }
    // reduce across the 16 lanes of each half-wave (masks <=8 stay in half)
#pragma unroll
    for (int m = 1; m < 16; m <<= 1) {
#pragma unroll
        for (int j = 0; j < 8; ++j) {
            den[j] += __shfl_xor(den[j], m, 32);
            num[j] += __shfl_xor(num[j], m, 32);
        }
    }
    float lsum = 0.f;
    if ((lane & 15) == 0) {
#pragma unroll
        for (int j = 0; j < 8; ++j) {
            const float ratio = (num[j] + EPS_F) / (den[j] + EPS_F) + EPS_F;
            lsum -= LN2_F * __builtin_amdgcn_logf(ratio);   // ln(x) = ln2*log2(x)
        }
    }
    lsum += __shfl_xor(lsum, 16, 32);          // lane0 += lane16 partial
    if (lane == 0) shWave[wave] = lsum;
    __syncthreads();
    if (tid == 0) {
        float s = 0.f;
#pragma unroll
        for (int w = 0; w < WAVES; ++w) s += shWave[w];
        partials[blockIdx.x] = s;
    }
}

// ---------------- kernel 3: deterministic final reduction -------------------
__global__ void __launch_bounds__(256) reduce_partials_kernel(
    const float* __restrict__ part, float* __restrict__ out)
{
    __shared__ float red[256];
    float s = 0.f;
    for (int i = threadIdx.x; i < NBLOCKS; i += 256) s += part[i];
    red[threadIdx.x] = s;
    __syncthreads();
    for (int off = 128; off > 0; off >>= 1) {
        if (threadIdx.x < off) red[threadIdx.x] += red[threadIdx.x + off];
        __syncthreads();
    }
    if (threadIdx.x == 0) out[0] = red[0] * (1.0f / (float)B_ROWS);
}

// ---------------------------------------------------------------------------
extern "C" void kernel_launch(void* const* d_in, const int* in_sizes, int n_in,
                              void* d_out, int out_size, void* d_ws, size_t ws_size,
                              hipStream_t stream)
{
    (void)in_sizes; (void)n_in; (void)out_size; (void)ws_size;
    const float* feats  = (const float*)d_in[0];
    const float* protos = (const float*)d_in[1];
    const int*   labels = (const int*)d_in[2];
    float* out = (float*)d_out;

    // workspace layout: [0, 32768): f16 normalized prototypes
    //                   [32768, 32768 + 2048*4): per-block partial loss sums
    _Float16* pnorm  = (_Float16*)d_ws;
    float* partials  = (float*)((char*)d_ws + (size_t)KPROTO * DDIM * sizeof(_Float16));

    proto_norm_kernel<<<KPROTO, DDIM, 0, stream>>>(protos, pnorm);
    protonce_main_kernel<<<NBLOCKS, 256, 0, stream>>>(feats, pnorm, labels, partials);
    reduce_partials_kernel<<<1, 256, 0, stream>>>(partials, out);
}
